// ResBlock_B_15814069584194
// MI455X (gfx1250) — compile-verified, matrix operand order checked by reading, not run on hardware
//
#include <hip/hip_runtime.h>

// CDNA5 / gfx1250 KPConv residual block, round 3.
// All contractions on V_WMMA_F32_16X16X32_F16 (f16 in, fp32 accumulate).
// Weights pre-packed into per-lane fragment order so every B fragment is a
// single 32-byte load (2x global_load_b128). Neighbor-feature gather staged
// through LDS via global_load_async_to_lds_b128. Geometry/BN/lrelu stay fp32.

typedef __attribute__((ext_vector_type(16))) _Float16 v16h;
typedef __attribute__((ext_vector_type(8)))  _Float16 v8h;
typedef __attribute__((ext_vector_type(8)))  float    v8f;
typedef __attribute__((ext_vector_type(4)))  float    v4f;
typedef int v4i __attribute__((vector_size(16)));   // matches async builtin's pointee

#define LRELU(x) ((x) >= 0.f ? (x) : 0.1f * (x))
#define BN_EPS 1e-5f
#define KP_INV_EXTENT 20.0f   // 1 / 0.05

#if defined(__has_builtin)
#if __has_builtin(__builtin_amdgcn_global_load_async_to_lds_b128)
#define HAVE_ASYNC_LDS 1
#endif
#endif

// K position inside a 16x16x32 f16 A/B fragment (ISA 7.12.2):
// lanes 0-15 hold K = {0..7, 16..23}, lanes 16-31 hold K = {8..15, 24..31},
// packed as (K, K+1) pairs per VGPR.
__device__ __forceinline__ int kpos(int v, int lane) {
    return ((v >> 2) << 4) + ((lane >> 4) << 3) + ((v & 3) << 1);
}

// ------------------------------------------------------------ weight packing
// Pack f32 weight W[K][C] into f16 B-fragment order:
// P[frag][lane][h], frag = (kt * (C/16) + j), h = 0..15, where the value is
// W[kt*32 + kpos(h>>1,lane) + (h&1)][j*16 + (lane&15)].
__global__ void pack_b_kernel(const float* __restrict__ W,
                              _Float16* __restrict__ P, int K, int C) {
    const int idx = blockIdx.x * 256 + threadIdx.x;
    if (idx >= K * C) return;
    const int h    = idx & 15;
    const int lane = (idx >> 4) & 31;
    const int f    = idx >> 9;
    const int CT   = C >> 4;
    const int kt   = f / CT;
    const int j    = f - kt * CT;
    const int k    = kt * 32 + kpos(h >> 1, lane) + (h & 1);
    const int col  = (j << 4) + (lane & 15);
    P[idx] = (_Float16)W[k * C + col];
}

// ------------------------------------------------------------ conv1 + BN + lrelu
// x = lrelu(bn1(features @ W1)) -> f16 [N,64].  One wave = 16 rows x 64 cols.
__global__ __launch_bounds__(128) void conv1_kernel(
    const float* __restrict__ feat,       // [N,128]
    const _Float16* __restrict__ W1p,     // packed, K=128 C=64
    const float* __restrict__ g, const float* __restrict__ bb,
    const float* __restrict__ bm, const float* __restrict__ bv,
    _Float16* __restrict__ xh, int N)
{
    const int lane = threadIdx.x & 31;
    const int tile = blockIdx.x * 4 + (threadIdx.x >> 5);
    const int row0 = tile * 16;
    if (row0 >= N) return;
    const int arow = row0 + (lane & 15);
    const int loff = (lane >> 4) << 3;

    __builtin_prefetch((const void*)(feat + (size_t)(arow + 16) * 128), 0, 0);

    v8f acc[4] = {};
    #pragma unroll
    for (int kk = 0; kk < 128; kk += 32) {
        v16h a;
        #pragma unroll
        for (int gI = 0; gI < 2; ++gI) {
            const float* p = feat + (size_t)arow * 128 + kk + gI * 16 + loff;
            const v4f f0 = *(const v4f*)p;
            const v4f f1 = *(const v4f*)(p + 4);
            #pragma unroll
            for (int e = 0; e < 4; ++e) {
                a[gI * 8 + e]     = (_Float16)f0[e];
                a[gI * 8 + 4 + e] = (_Float16)f1[e];
            }
        }
        #pragma unroll
        for (int j = 0; j < 4; ++j) {
            const v16h b = *(const v16h*)(W1p + ((size_t)(((kk >> 5) << 2) + j) << 9) + (lane << 4));
            acc[j] = __builtin_amdgcn_wmma_f32_16x16x32_f16(
                false, a, false, b, (short)0, acc[j], false, false);
        }
    }
    #pragma unroll
    for (int j = 0; j < 4; ++j) {
        const int col = (j << 4) + (lane & 15);
        const float sc = g[col] * rsqrtf(bv[col] + BN_EPS);
        const float bi = bb[col] - bm[col] * sc;
        #pragma unroll
        for (int r = 0; r < 8; ++r) {
            const int row = row0 + r + ((lane >> 4) << 3);
            float val = fmaf(acc[j][r], sc, bi);
            val = LRELU(val);
            xh[(size_t)row * 64 + col] = (_Float16)val;
        }
    }
}

// ------------------------------------------------------------ KPConv
// One wave per 16-point tile.
// Per point: stage 32 neighbor feature rows (f16, 128B each) into LDS
// (async-to-LDS b128), build influence fragment in fp32,
// wf(16x64) = h^T(16x32) @ f_n(32x64) -> LDS as f16.
// Then Y(16x64) = WF(16x960) @ Wkp_packed, lrelu -> y f16.
__global__ __launch_bounds__(32) void kpconv_kernel(
    const _Float16* __restrict__ xh,     // [N,64]
    const float* __restrict__ points,    // [N,3]
    const int* __restrict__ nidx,        // [N,32]
    const float* __restrict__ kpts,      // [15,3]
    const _Float16* __restrict__ Wkpp,   // packed, K=960 C=64
    _Float16* __restrict__ yh, int N)
{
    __shared__ __align__(16) float    dvec[32][4];
    __shared__ __align__(16) _Float16 fLDS[32][72];   // 32 neighbor rows, padded
    __shared__ __align__(16) _Float16 wf[16][960];

    const int lane = threadIdx.x;
    const int n0 = blockIdx.x * 16;
    if (n0 >= N) return;
    const int prow = lane & 15;
    const int loff = (lane >> 4) << 3;

    float kx = 0.f, ky = 0.f, kz = 0.f;
    if (prow < 15) {
        kx = kpts[prow * 3 + 0];
        ky = kpts[prow * 3 + 1];
        kz = kpts[prow * 3 + 2];
    }

    for (int pt = 0; pt < 16; ++pt) {
        const int n = n0 + pt;
        const int id = nidx[(size_t)n * 32 + lane];   // this lane's neighbor
        const float px = points[n * 3 + 0];
        const float py = points[n * 3 + 1];
        const float pz = points[n * 3 + 2];
        dvec[lane][0] = points[id * 3 + 0] - px;
        dvec[lane][1] = points[id * 3 + 1] - py;
        dvec[lane][2] = points[id * 3 + 2] - pz;

        // stage this lane's neighbor feature row (64 halves = 128B) into LDS
        const _Float16* src = xh + (size_t)id * 64;
#ifdef HAVE_ASYNC_LDS
        #pragma unroll
        for (int c = 0; c < 8; ++c) {
            __builtin_amdgcn_global_load_async_to_lds_b128(
                (v4i*)(src + c * 8),
                (v4i*)(&fLDS[lane][c * 8]),
                0, 0);
        }
#if __has_builtin(__builtin_amdgcn_s_wait_asynccnt)
        __builtin_amdgcn_s_wait_asynccnt(0);
#else
        asm volatile("s_wait_asynccnt 0" ::: "memory");
#endif
#else
        #pragma unroll
        for (int c = 0; c < 8; ++c)
            *(v8h*)(&fLDS[lane][c * 8]) = *(const v8h*)(src + c * 8);
#endif
        __syncthreads();

        // A = h^T : A[p][k] = max(0, 1 - |d_k - kp_p| / 0.05), row 15 = 0 pad
        v16h a;
        #pragma unroll
        for (int v = 0; v < 8; ++v) {
            const int kb = kpos(v, lane);
            #pragma unroll
            for (int t = 0; t < 2; ++t) {
                const int k = kb + t;
                const float dx = dvec[k][0] - kx;
                const float dy = dvec[k][1] - ky;
                const float dz = dvec[k][2] - kz;
                const float dist = sqrtf(dx * dx + dy * dy + dz * dz);
                float h = fmaxf(0.f, 1.f - dist * KP_INV_EXTENT);
                if (prow >= 15) h = 0.f;
                a[2 * v + t] = (_Float16)h;
            }
        }
        // B = staged neighbor features, 4 column tiles, from LDS
        #pragma unroll
        for (int j = 0; j < 4; ++j) {
            const int col = (j << 4) + (lane & 15);
            v16h b;
            #pragma unroll
            for (int v = 0; v < 8; ++v) {
                const int kb = kpos(v, lane);
                b[2 * v]     = fLDS[kb][col];
                b[2 * v + 1] = fLDS[kb + 1][col];
            }
            v8f c = {};
            c = __builtin_amdgcn_wmma_f32_16x16x32_f16(
                false, a, false, b, (short)0, c, false, false);
            // scatter C into WF row `pt`: column index = p*64 + c
            #pragma unroll
            for (int r = 0; r < 8; ++r) {
                const int p = r + ((lane >> 4) << 3);
                wf[pt][p * 64 + (j << 4) + (lane & 15)] = (_Float16)c[r];
            }
        }
        __syncthreads();
    }

    // Stage B: Y = WF @ Wkp_flat  (K = 960 = 30 k-steps)
    v8f acc[4] = {};
    const int row = lane & 15;
    for (int kk = 0; kk < 960; kk += 32) {
        const v8h lo = *(const v8h*)(&wf[row][kk + loff]);
        const v8h hi = *(const v8h*)(&wf[row][kk + 16 + loff]);
        const v16h a = __builtin_shufflevector(lo, hi, 0, 1, 2, 3, 4, 5, 6, 7,
                                               8, 9, 10, 11, 12, 13, 14, 15);
        #pragma unroll
        for (int j = 0; j < 4; ++j) {
            const v16h b = *(const v16h*)(Wkpp + ((size_t)(((kk >> 5) << 2) + j) << 9) + (lane << 4));
            acc[j] = __builtin_amdgcn_wmma_f32_16x16x32_f16(
                false, a, false, b, (short)0, acc[j], false, false);
        }
    }
    #pragma unroll
    for (int j = 0; j < 4; ++j) {
        const int col = (j << 4) + (lane & 15);
        #pragma unroll
        for (int r = 0; r < 8; ++r) {
            const int orow = n0 + r + ((lane >> 4) << 3);
            float val = acc[j][r];
            val = LRELU(val);
            yh[(size_t)orow * 64 + col] = (_Float16)val;
        }
    }
}

// ------------------------------------------------------------ conv2 + shortcut + residual
// out = lrelu( lrelu(bn2(y@W2)) + lrelu(bns(features@Ws)) )
// Block = 4 waves; wave w owns column group w (64 of 256 cols) of a 16-row tile.
__global__ __launch_bounds__(128) void conv2_res_kernel(
    const _Float16* __restrict__ yh,    // [N,64]
    const _Float16* __restrict__ W2p,   // packed, K=64 C=256
    const float* __restrict__ g2, const float* __restrict__ b2,
    const float* __restrict__ m2, const float* __restrict__ v2p,
    const float* __restrict__ feat,     // [N,128]
    const _Float16* __restrict__ Wsp,   // packed, K=128 C=256
    const float* __restrict__ gs, const float* __restrict__ bsp,
    const float* __restrict__ ms, const float* __restrict__ vsp,
    float* __restrict__ out, int N)
{
    const int lane = threadIdx.x & 31;
    const int cg = threadIdx.x >> 5;          // column group 0..3
    const int row0 = blockIdx.x * 16;
    if (row0 >= N) return;
    const int arow = row0 + (lane & 15);
    const int loff = (lane >> 4) << 3;

    v8f accy[4] = {};
    v8f accs[4] = {};

    // y @ W2 : K = 64
    #pragma unroll
    for (int kk = 0; kk < 64; kk += 32) {
        const v8h lo = *(const v8h*)(yh + (size_t)arow * 64 + kk + loff);
        const v8h hi = *(const v8h*)(yh + (size_t)arow * 64 + kk + 16 + loff);
        const v16h a = __builtin_shufflevector(lo, hi, 0, 1, 2, 3, 4, 5, 6, 7,
                                               8, 9, 10, 11, 12, 13, 14, 15);
        #pragma unroll
        for (int j = 0; j < 4; ++j) {
            const int frag = ((kk >> 5) << 4) + (cg << 2) + j;
            const v16h b = *(const v16h*)(W2p + ((size_t)frag << 9) + (lane << 4));
            accy[j] = __builtin_amdgcn_wmma_f32_16x16x32_f16(
                false, a, false, b, (short)0, accy[j], false, false);
        }
    }
    // features @ Ws : K = 128
    #pragma unroll
    for (int kk = 0; kk < 128; kk += 32) {
        v16h a;
        #pragma unroll
        for (int gI = 0; gI < 2; ++gI) {
            const float* p = feat + (size_t)arow * 128 + kk + gI * 16 + loff;
            const v4f f0 = *(const v4f*)p;
            const v4f f1 = *(const v4f*)(p + 4);
            #pragma unroll
            for (int e = 0; e < 4; ++e) {
                a[gI * 8 + e]     = (_Float16)f0[e];
                a[gI * 8 + 4 + e] = (_Float16)f1[e];
            }
        }
        #pragma unroll
        for (int j = 0; j < 4; ++j) {
            const int frag = ((kk >> 5) << 4) + (cg << 2) + j;
            const v16h b = *(const v16h*)(Wsp + ((size_t)frag << 9) + (lane << 4));
            accs[j] = __builtin_amdgcn_wmma_f32_16x16x32_f16(
                false, a, false, b, (short)0, accs[j], false, false);
        }
    }
    // epilogue: BN both paths, lrelu each, add, final lrelu
    #pragma unroll
    for (int j = 0; j < 4; ++j) {
        const int col = (cg << 6) + (j << 4) + (lane & 15);
        const float s2 = g2[col] * rsqrtf(v2p[col] + BN_EPS);
        const float o2 = b2[col] - m2[col] * s2;
        const float ss = gs[col] * rsqrtf(vsp[col] + BN_EPS);
        const float os = bsp[col] - ms[col] * ss;
        #pragma unroll
        for (int r = 0; r < 8; ++r) {
            const int orow = row0 + r + ((lane >> 4) << 3);
            float vy = fmaf(accy[j][r], s2, o2);
            vy = LRELU(vy);
            float vsh = fmaf(accs[j][r], ss, os);
            vsh = LRELU(vsh);
            float o = vy + vsh;
            o = LRELU(o);
            out[(size_t)orow * 256 + col] = o;
        }
    }
}

// ------------------------------------------------------------ launch
extern "C" void kernel_launch(void* const* d_in, const int* in_sizes, int n_in,
                              void* d_out, int out_size, void* d_ws, size_t ws_size,
                              hipStream_t stream) {
    const float* feat   = (const float*)d_in[0];
    const float* points = (const float*)d_in[1];
    const int*   nidx   = (const int*)d_in[2];
    const float* W1     = (const float*)d_in[3];
    const float* bn1_g  = (const float*)d_in[4];
    const float* bn1_b  = (const float*)d_in[5];
    const float* bn1_m  = (const float*)d_in[6];
    const float* bn1_v  = (const float*)d_in[7];
    const float* kpts   = (const float*)d_in[8];
    const float* Wkp    = (const float*)d_in[9];
    const float* W2     = (const float*)d_in[10];
    const float* bn2_g  = (const float*)d_in[11];
    const float* bn2_b  = (const float*)d_in[12];
    const float* bn2_m  = (const float*)d_in[13];
    const float* bn2_v  = (const float*)d_in[14];
    const float* Ws     = (const float*)d_in[15];
    const float* bns_g  = (const float*)d_in[16];
    const float* bns_b  = (const float*)d_in[17];
    const float* bns_m  = (const float*)d_in[18];
    const float* bns_v  = (const float*)d_in[19];
    float* out = (float*)d_out;

    const int N = in_sizes[0] / 128;   // 40000 (multiple of 16)
    const int tiles = N / 16;          // 2500

    // workspace: packed f16 weights + f16 intermediates (all 32B aligned)
    _Float16* W1p  = (_Float16*)d_ws;                 // 128*64   = 8192
    _Float16* W2p  = W1p + 128 * 64;                  // 64*256   = 16384
    _Float16* Wsp  = W2p + 64 * 256;                  // 128*256  = 32768
    _Float16* Wkpp = Wsp + 128 * 256;                 // 960*64   = 61440
    _Float16* xh   = Wkpp + 960 * 64;                 // N*64
    _Float16* yh   = xh + (size_t)N * 64;             // N*64

    pack_b_kernel<<<(128 * 64 + 255) / 256, 256, 0, stream>>>(W1, W1p, 128, 64);
    pack_b_kernel<<<(64 * 256 + 255) / 256, 256, 0, stream>>>(W2, W2p, 64, 256);
    pack_b_kernel<<<(128 * 256 + 255) / 256, 256, 0, stream>>>(Ws, Wsp, 128, 256);
    pack_b_kernel<<<(960 * 64 + 255) / 256, 256, 0, stream>>>(Wkp, Wkpp, 960, 64);

    conv1_kernel<<<(tiles + 3) / 4, 128, 0, stream>>>(
        feat, W1p, bn1_g, bn1_b, bn1_m, bn1_v, xh, N);

    kpconv_kernel<<<tiles, 32, 0, stream>>>(
        xh, points, nidx, kpts, Wkpp, yh, N);

    conv2_res_kernel<<<tiles, 128, 0, stream>>>(
        yh, W2p, bn2_g, bn2_b, bn2_m, bn2_v,
        feat, Wsp, bns_g, bns_b, bns_m, bns_v, out, N);
}
